// RWKVRegionCell_67860483276967
// MI455X (gfx1250) — matrix operand order repaired and checked by reading, not compiled, not run on hardware
//
#include <hip/hip_runtime.h>
#include <hip/hip_bf16.h>
#include <math.h>

// ---------------------------------------------------------------------------
// Problem constants (from reference)
// ---------------------------------------------------------------------------
#define BB   64
#define TT   256
#define DD   1024
#define MROWS (BB * TT)      // 16384 rows of the batched GEMMs
#define MPAIR 16             // RoPE frequency pairs
#define KWTA 128             // top-k kept per row

// LDS A-tile layout: 16 rows x 1024 bf16, TDM pad of 16B after every 1024B
// -> effective row stride 2080 B (lanes 8 banks apart, all chunks 16B aligned)
#define ROW_STRIDE 2080
#define ATILE_BYTES (16 * ROW_STRIDE)

typedef __attribute__((ext_vector_type(16))) __bf16 v16bf;
typedef __attribute__((ext_vector_type(8)))  float  v8f;
typedef __attribute__((ext_vector_type(4)))  unsigned int u32x4;
typedef __attribute__((ext_vector_type(8)))  int    i32x8;
typedef __attribute__((ext_vector_type(4)))  int    i32x4;

union AFrag { v16bf v; u32x4 u[2]; };   // 32 bytes: 16 bf16 lanes-worth

#if defined(__has_builtin)
#if __has_builtin(__builtin_amdgcn_tensor_load_to_lds) && \
    __has_builtin(__builtin_amdgcn_s_wait_tensorcnt)
#define HAVE_TDM 1
#endif
#endif

__device__ __forceinline__ unsigned short f2bf(float f) {
    unsigned int u = __float_as_uint(f);
    unsigned int r = 0x7FFFu + ((u >> 16) & 1u);   // round-to-nearest-even
    return (unsigned short)((u + r) >> 16);
}
__device__ __forceinline__ float sigmoidf(float x) { return 1.0f / (1.0f + expf(-x)); }

// B fragment: lane half 0 -> K{0..7,16..23}, half 1 -> +8 (matches A)
__device__ __forceinline__ void loadB(const unsigned short* __restrict__ Wrow,
                                      int kt, int half, AFrag& b) {
    const int cb = kt * 32 + half * 8;
    b.u[0] = *(const u32x4*)(Wrow + cb);
    b.u[1] = *(const u32x4*)(Wrow + cb + 16);
}

// A fragment from padded LDS tile: lane half 0 -> K{0..7,16..23}, half 1 -> +8
__device__ __forceinline__ void loadA_lds(const unsigned char* At, int lm,
                                          int kt, int half, AFrag& a) {
    const int ca = kt * 32 + half * 8;         // element column of chunk 0
    const int c2 = ca + 16;                    // element column of chunk 1
    const unsigned o1 = (unsigned)(lm * ROW_STRIDE + 2 * ca + ((ca >= 512) ? 16 : 0));
    const unsigned o2 = (unsigned)(lm * ROW_STRIDE + 2 * c2 + ((c2 >= 512) ? 16 : 0));
    a.u[0] = *(const u32x4*)(At + o1);
    a.u[1] = *(const u32x4*)(At + o2);
}

// Stage one 16x1024 bf16 tile into LDS (TDM if available, else cooperative).
// Must be called by ALL threads of the block; ends with the tile visible.
#define STAGE_ATILE(SRC_PTR, SH_TILE)                                           \
    do {                                                                        \
        stage_tile_body((SRC_PTR), (SH_TILE));                                  \
        __syncthreads();                                                        \
    } while (0)

__device__ __forceinline__ void stage_tile_body(const unsigned short* __restrict__ src,
                                                unsigned char* sh) {
#if defined(HAVE_TDM)
    if (threadIdx.x < 32) {   // wave 0 issues one TDM op for the block
        const unsigned long long ga = (unsigned long long)(size_t)src;
        const unsigned lds = (unsigned)(size_t)(void*)sh;
        u32x4 g0;
        g0[0] = 1u;                                   // count=1, user mode, no gather
        g0[1] = lds;                                  // LDS byte address
        g0[2] = (unsigned)(ga & 0xFFFFFFFFu);         // global_addr[31:0]
        g0[3] = (unsigned)((ga >> 32) & 0x01FFFFFFu)  // global_addr[56:32]
              | (2u << 30);                           // type = 2 (image)
        i32x8 g1;
        g1[0] = (1 << 16)                             // data_size = 1 (2 bytes)
              | (1 << 20)                             // pad_enable
              | (7 << 22)                             // pad_interval: 256 DWORDs
              | (3 << 25);                            // pad_amount: 4 DWORDs (16B)
        g1[1] = (int)((DD & 0xFFFF) << 16);           // tensor_dim0[15:0] @ bits 63:48
        g1[2] = (int)(((DD >> 16) & 0xFFFF)           // tensor_dim0[31:16]
              | (16 << 16));                          // tensor_dim1[15:0] = 16 rows
        g1[3] = (int)((DD & 0xFFFF) << 16);           // tile_dim0 = 1024 @ bits 127:112
        g1[4] = 16;                                   // tile_dim1 = 16 (tile_dim2 = 0)
        g1[5] = DD;                                   // tensor_dim0_stride[31:0] = 1024
        g1[6] = 0;                                    // stride hi + dim1_stride lo
        g1[7] = 0;
        const i32x4 z4 = {0, 0, 0, 0};
        const i32x8 z8 = {0, 0, 0, 0, 0, 0, 0, 0};
        // clang-23 / therock form: (g0, g1, g2, g3, g4, cpol)
        __builtin_amdgcn_tensor_load_to_lds(g0, g1, z4, z4, z8, 0);
        __builtin_amdgcn_s_wait_tensorcnt(0);
    }
#else
    // Fallback: cooperative copy with the same padded layout.
    for (int idx = threadIdx.x; idx < 16 * 128; idx += 256) {
        const int row = idx >> 7;
        const int chk = idx & 127;                        // 16B chunk within row
        const unsigned off = (unsigned)(row * ROW_STRIDE + chk * 16 +
                                        ((chk >= 64) ? 16 : 0));
        *(u32x4*)(sh + off) = *(const u32x4*)(src + (size_t)row * DD + chk * 8);
    }
#endif
}

// ---------------------------------------------------------------------------
// 0) fp32 -> bf16 weight conversion (Wr, Wk, Wv, Wo), 8 MB total -> lives in L2
// ---------------------------------------------------------------------------
__global__ __launch_bounds__(256) void wconv_kernel(
    const float* __restrict__ wr, const float* __restrict__ wk,
    const float* __restrict__ wv, const float* __restrict__ wo,
    unsigned short* __restrict__ wrb, unsigned short* __restrict__ wkb,
    unsigned short* __restrict__ wvb, unsigned short* __restrict__ wob) {
    int i = blockIdx.x * 256 + threadIdx.x;       // DD*DD threads
    wrb[i] = f2bf(wr[i]);
    wkb[i] = f2bf(wk[i]);
    wvb[i] = f2bf(wv[i]);
    wob[i] = f2bf(wo[i]);
}

// ---------------------------------------------------------------------------
// 1) rmsnorm: one block per row (B*T rows), 256 thr x 4 elems
// ---------------------------------------------------------------------------
__global__ __launch_bounds__(256) void rmsnorm_kernel(
    const float* __restrict__ X, const float* __restrict__ nw,
    float* __restrict__ XN, unsigned short* __restrict__ XNb) {
    __shared__ float sred[256];
    const int row = blockIdx.x, tid = threadIdx.x;
    const float4 xv = *((const float4*)(X + (size_t)row * DD) + tid);
    sred[tid] = xv.x * xv.x + xv.y * xv.y + xv.z * xv.z + xv.w * xv.w;
    __syncthreads();
    for (int s = 128; s > 0; s >>= 1) {
        if (tid < s) sred[tid] += sred[tid + s];
        __syncthreads();
    }
    const float inv = 1.0f / sqrtf(sred[0] * (1.0f / DD) + 1e-5f);
    const float4 wv = *((const float4*)nw + tid);
    float4 o;
    o.x = wv.x * xv.x * inv; o.y = wv.y * xv.y * inv;
    o.z = wv.z * xv.z * inv; o.w = wv.w * xv.w * inv;
    *((float4*)(XN + (size_t)row * DD) + tid) = o;
    ushort4 ob = make_ushort4(f2bf(o.x), f2bf(o.y), f2bf(o.z), f2bf(o.w));
    *((ushort4*)(XNb + (size_t)row * DD) + tid) = ob;
}

// ---------------------------------------------------------------------------
// 2) Fused RKV GEMM. A-tile staged to LDS by TDM (shared by 8 waves),
//    B fragments double-buffered in registers to overlap loads with WMMA.
// ---------------------------------------------------------------------------
__global__ __launch_bounds__(256) void gemm_rkv_kernel(
    const unsigned short* __restrict__ Abf,
    const unsigned short* __restrict__ Wr, const unsigned short* __restrict__ Wk,
    const unsigned short* __restrict__ Wv,
    float* __restrict__ R, float* __restrict__ Wx, float* __restrict__ V) {
    __shared__ unsigned char Atile[ATILE_BYTES];
    const int lane = threadIdx.x & 31;
    const int wave = threadIdx.x >> 5;
    const int half = lane >> 4;
    const int lm   = lane & 15;
    const int m_tile = blockIdx.x;                 // 0..1023
    const int n_tile = blockIdx.y * 8 + wave;      // 0..63

    STAGE_ATILE(Abf + (size_t)m_tile * 16 * DD, Atile);

    const unsigned short* BrR = Wr + (size_t)(n_tile * 16 + lm) * DD;
    const unsigned short* BkR = Wk + (size_t)(n_tile * 16 + lm) * DD;
    const unsigned short* BvR = Wv + (size_t)(n_tile * 16 + lm) * DD;

    AFrag a, br[2], bk[2], bv[2];
    v8f accR = {}, accK = {}, accV = {};

    loadB(BrR, 0, half, br[0]);
    loadB(BkR, 0, half, bk[0]);
    loadB(BvR, 0, half, bv[0]);
#pragma unroll
    for (int kt = 0; kt < DD / 32; ++kt) {
        const int cur = kt & 1, nxt = cur ^ 1;
        if (kt < DD / 32 - 1) {       // prefetch next k-tile weights
            loadB(BrR, kt + 1, half, br[nxt]);
            loadB(BkR, kt + 1, half, bk[nxt]);
            loadB(BvR, kt + 1, half, bv[nxt]);
        }
        loadA_lds(Atile, lm, kt, half, a);
        accR = __builtin_amdgcn_wmma_f32_16x16x32_bf16(false, a.v, false, br[cur].v,
                                                       (short)0, accR, false, false);
        accK = __builtin_amdgcn_wmma_f32_16x16x32_bf16(false, a.v, false, bk[cur].v,
                                                       (short)0, accK, false, false);
        accV = __builtin_amdgcn_wmma_f32_16x16x32_bf16(false, a.v, false, bv[cur].v,
                                                       (short)0, accV, false, false);
    }

    // D layout: VGPR r holds M = half*8 + r, N = lm
#pragma unroll
    for (int r = 0; r < 8; ++r) {
        const int m = m_tile * 16 + half * 8 + r;
        const int n = n_tile * 16 + lm;
        const size_t o = (size_t)m * DD + n;
        R[o]  = sigmoidf(accR[r]);
        Wx[o] = expf(accK[r]);
        V[o]  = accV[r];
    }
}

// ---------------------------------------------------------------------------
// 3) Scan: one thread per (b,d) channel, sequential over T.
// ---------------------------------------------------------------------------
__global__ __launch_bounds__(256) void scan_kernel(
    const float* __restrict__ R, const float* __restrict__ Wx,
    const float* __restrict__ V, const float* __restrict__ step_pos,
    const float* __restrict__ decay, unsigned short* __restrict__ Yb) {
    const int g = blockIdx.x * 256 + threadIdx.x;  // 0 .. B*D-1
    const int b = g >> 10;
    const int d = g & (DD - 1);
    const float sg  = sigmoidf(decay[d]);
    const float lam = sg * sg;
    const bool rope = d < 2 * MPAIR;
    float wt = 0.0f;
    if (rope) {  // W_time[j] = exp(log(1/1e4) * (1 - j/(M-1)))
        const int j = d >> 1;
        wt = expf(logf(1.0f / 10000.0f) * (1.0f - (float)j / (float)(MPAIR - 1)));
    }
    float num = 0.0f, den = 0.0f;
    for (int t = 0; t < TT; ++t) {
        const size_t o = ((size_t)(b * TT + t)) * DD + d;
        const float w = Wx[o];
        float v = V[o];
        if (rope) {
            const float th = step_pos[t] * wt;
            const float c = cosf(th), s = sinf(th);
            const float vp = V[o ^ 1];   // partner channel (d^1) in same row
            v = (d & 1) ? (vp * s + v * c) : (v * c - vp * s);
        }
        num = num * lam + w * v;
        den = den * lam + w;
        const float y = R[o] * (num / (den + 1e-9f));
        Yb[o] = f2bf(y);
    }
}

// ---------------------------------------------------------------------------
// 4) Wo GEMM with fused residual: H = XN + Y @ Wo^T  (writes d_out directly)
// ---------------------------------------------------------------------------
__global__ __launch_bounds__(256) void gemm_o_kernel(
    const unsigned short* __restrict__ Yb, const unsigned short* __restrict__ Wo,
    const float* __restrict__ XN, float* __restrict__ H) {
    __shared__ unsigned char Atile[ATILE_BYTES];
    const int lane = threadIdx.x & 31;
    const int wave = threadIdx.x >> 5;
    const int half = lane >> 4;
    const int lm   = lane & 15;
    const int m_tile = blockIdx.x;
    const int n_tile = blockIdx.y * 8 + wave;

    STAGE_ATILE(Yb + (size_t)m_tile * 16 * DD, Atile);

    const unsigned short* BoR = Wo + (size_t)(n_tile * 16 + lm) * DD;

    AFrag a, bo[2];
    v8f acc = {};
    loadB(BoR, 0, half, bo[0]);
#pragma unroll
    for (int kt = 0; kt < DD / 32; ++kt) {
        const int cur = kt & 1, nxt = cur ^ 1;
        if (kt < DD / 32 - 1) loadB(BoR, kt + 1, half, bo[nxt]);
        loadA_lds(Atile, lm, kt, half, a);
        acc = __builtin_amdgcn_wmma_f32_16x16x32_bf16(false, a.v, false, bo[cur].v,
                                                      (short)0, acc, false, false);
    }
#pragma unroll
    for (int r = 0; r < 8; ++r) {
        const int m = m_tile * 16 + half * 8 + r;
        const int n = n_tile * 16 + lm;
        const size_t o = (size_t)m * DD + n;
        H[o] = XN[o] + acc[r];
    }
}

// ---------------------------------------------------------------------------
// 5) KWTA (in-place on d_out): per row, keep top-128 by |h|.
// ---------------------------------------------------------------------------
__global__ __launch_bounds__(256) void kwta_kernel(float* __restrict__ H) {
    __shared__ float sred[256];
    __shared__ int   scnt[256];
    const int row = blockIdx.x, tid = threadIdx.x;
    float* hrow = H + (size_t)row * DD;
    float4 hv = *((const float4*)hrow + tid);
    const float a0 = fabsf(hv.x), a1 = fabsf(hv.y);
    const float a2 = fabsf(hv.z), a3 = fabsf(hv.w);
    sred[tid] = fmaxf(fmaxf(a0, a1), fmaxf(a2, a3));
    __syncthreads();
    for (int s = 128; s > 0; s >>= 1) {
        if (tid < s) sred[tid] = fmaxf(sred[tid], sred[tid + s]);
        __syncthreads();
    }
    float lo = 0.0f, hi = sred[0];
    for (int it = 0; it < 24; ++it) {
        const float mid = 0.5f * (lo + hi);
        scnt[tid] = (a0 >= mid) + (a1 >= mid) + (a2 >= mid) + (a3 >= mid);
        __syncthreads();
        for (int s = 128; s > 0; s >>= 1) {
            if (tid < s) scnt[tid] += scnt[tid + s];
            __syncthreads();
        }
        const int total = scnt[0];
        __syncthreads();
        if (total >= KWTA) lo = mid; else hi = mid;   // uniform branch
    }
    hv.x = (a0 >= lo) ? hv.x : 0.0f;
    hv.y = (a1 >= lo) ? hv.y : 0.0f;
    hv.z = (a2 >= lo) ? hv.z : 0.0f;
    hv.w = (a3 >= lo) ? hv.w : 0.0f;
    *((float4*)hrow + tid) = hv;
}

// ---------------------------------------------------------------------------
// Launch
// ---------------------------------------------------------------------------
extern "C" void kernel_launch(void* const* d_in, const int* in_sizes, int n_in,
                              void* d_out, int out_size, void* d_ws, size_t ws_size,
                              hipStream_t stream) {
    const float* x        = (const float*)d_in[0];
    const float* step_pos = (const float*)d_in[1];
    const float* norm_w   = (const float*)d_in[2];
    const float* Wr       = (const float*)d_in[3];
    const float* Wk       = (const float*)d_in[4];
    const float* Wv       = (const float*)d_in[5];
    const float* Wo       = (const float*)d_in[6];
    const float* decay    = (const float*)d_in[7];
    float* out = (float*)d_out;

    // workspace layout
    char* base = (char*)d_ws;
    size_t off = 0;
    auto alloc = [&](size_t bytes) {
        size_t o = off;
        off += (bytes + 255) & ~(size_t)255;
        return o;
    };
    const size_t nMD = (size_t)MROWS * DD;
    float*          XN  = (float*)(base + alloc(nMD * 4));
    unsigned short* XNb = (unsigned short*)(base + alloc(nMD * 2));
    float*          R   = (float*)(base + alloc(nMD * 4));
    float*          Wx  = (float*)(base + alloc(nMD * 4));
    float*          V   = (float*)(base + alloc(nMD * 4));
    unsigned short* Yb  = (unsigned short*)(base + alloc(nMD * 2));
    unsigned short* Wrb = (unsigned short*)(base + alloc((size_t)DD * DD * 2));
    unsigned short* Wkb = (unsigned short*)(base + alloc((size_t)DD * DD * 2));
    unsigned short* Wvb = (unsigned short*)(base + alloc((size_t)DD * DD * 2));
    unsigned short* Wob = (unsigned short*)(base + alloc((size_t)DD * DD * 2));

    // 0) weights -> bf16
    wconv_kernel<<<(DD * DD) / 256, 256, 0, stream>>>(Wr, Wk, Wv, Wo,
                                                      Wrb, Wkb, Wvb, Wob);
    // 1) rmsnorm
    rmsnorm_kernel<<<MROWS, 256, 0, stream>>>(x, norm_w, XN, XNb);
    // 2) fused RKV GEMM (WMMA bf16, TDM-staged A tile)
    gemm_rkv_kernel<<<dim3(MROWS / 16, 8), 256, 0, stream>>>(XNb, Wrb, Wkb, Wvb,
                                                             R, Wx, V);
    // 3) per-channel scan (RoPE + EMA + gate)
    scan_kernel<<<(BB * DD) / 256, 256, 0, stream>>>(R, Wx, V, step_pos, decay, Yb);
    // 4) output GEMM with residual (WMMA bf16) -> d_out
    gemm_o_kernel<<<dim3(MROWS / 16, 8), 256, 0, stream>>>(Yb, Wob, XN, out);
    // 5) KWTA in place on d_out
    kwta_kernel<<<MROWS, 256, 0, stream>>>(out);
}
